// CFGSubASTExpressionCombiner_58274116272163
// MI455X (gfx1250) — matrix operand order repaired and checked by reading, not compiled, not run on hardware
//
#include <hip/hip_runtime.h>
#include <hip/hip_bf16.h>

typedef __attribute__((ext_vector_type(16))) __bf16 v16bf;
typedef __attribute__((ext_vector_type(8)))  float  v8f;

union Frag16 { v16bf v; uint4 q[2]; };
union U4     { uint4 q; unsigned short s[8]; };

__device__ __forceinline__ unsigned short f2bf(float f) {
    unsigned u = __float_as_uint(f);
    unsigned r = (u + 0x7fffu + ((u >> 16) & 1u)) >> 16;   // RNE
    return (unsigned short)r;
}
__device__ __forceinline__ float bf2f(unsigned short h) {
    return __uint_as_float(((unsigned)h) << 16);
}
// order-preserving float<->uint encoding for atomicMax
__device__ __forceinline__ unsigned ordf(float f) {
    unsigned u = __float_as_uint(f);
    return (u & 0x80000000u) ? ~u : (u | 0x80000000u);
}
__device__ __forceinline__ float unordf(unsigned e) {
    unsigned u = (e & 0x80000000u) ? (e & 0x7fffffffu) : ~e;
    return __uint_as_float(u);
}
#define ORD_NEG_INF 0x007FFFFFu   // ordf(-inf)

// ---------------------------------------------------------------------------
// Weight convert: Wt[n*256 + k] = bf16(W[k*256 + n])   (transpose to N-major)
__global__ void wconv_kernel(const float* __restrict__ W, unsigned short* __restrict__ Wt) {
    int i = blockIdx.x * blockDim.x + threadIdx.x;   // 65536 threads
    int k = i >> 8, n = i & 255;
    Wt[n * 256 + k] = f2bf(W[k * 256 + n]);
}

// init: pooled=0, attnq=0, segmax=ordf(-inf), denom=0
__global__ void init_kernel(float* __restrict__ pooled, unsigned short* __restrict__ attnq,
                            unsigned* __restrict__ segmax, float* __restrict__ denom, int S) {
    long long i = (long long)blockIdx.x * blockDim.x + threadIdx.x;
    long long total = (long long)S * 256;
    if (i < total) { pooled[i] = 0.0f; attnq[i] = 0; }
    if (i < (long long)S * 8) { segmax[i] = ORD_NEG_INF; denom[i] = 0.0f; }
}

// xb[m,:] = bf16(enc[mapk[m],:])
__global__ void gather_x_kernel(const float* __restrict__ enc, const int* __restrict__ key,
                                unsigned short* __restrict__ xb) {
    int m = blockIdx.x, d = threadIdx.x;
    xb[(size_t)m * 256 + d] = f2bf(enc[(size_t)key[m] * 256 + d]);
}

// attnq[pk[i],:] = bf16(enc[pv[i],:])
__global__ void scatter_q_kernel(const float* __restrict__ enc, const int* __restrict__ pk,
                                 const int* __restrict__ pv, unsigned short* __restrict__ attnq) {
    int i = blockIdx.x, d = threadIdx.x;
    attnq[(size_t)pk[i] * 256 + d] = f2bf(enc[(size_t)pv[i] * 256 + d]);
}

// ---------------------------------------------------------------------------
// Generic [rows,256] @ [256,256]^T(bf16, N-major) + bias GEMM via WMMA bf16.
// Block = 128 threads (4 waves). Wave w -> 16 rows x 64 cols (4 16x16 tiles).
template <int OUT_F32>
__global__ __launch_bounds__(128) void gemm256_bf16(
    const unsigned short* __restrict__ A, const unsigned short* __restrict__ Bt,
    const float* __restrict__ bias, float* __restrict__ outF,
    unsigned short* __restrict__ outH, int rows)
{
    const int lane = threadIdx.x & 31;
    const int wave = threadIdx.x >> 5;
    const int l15  = lane & 15;
    const int hi   = lane >> 4;            // lane-half selects K sub-block
    const int row0 = blockIdx.x * 16;
    const int col0 = wave * 64;

    int mload = row0 + l15;
    if (mload >= rows) mload = rows - 1;
    const uint4* Ar = (const uint4*)(A + (size_t)mload * 256);

    v8f acc[4] = {};
    for (int kk = 0; kk < 8; ++kk) {
        const int k = kk * 32;
        Frag16 a;
        // A 16x32 bf16, contiguous k order used here (matches B): lanes 0-15 K=[0..15]; lanes 16-31 K=[16..31]
        a.q[0] = Ar[(k >> 3) + 2 * hi];
        a.q[1] = Ar[(k >> 3) + 2 * hi + 1];
        #pragma unroll
        for (int t = 0; t < 4; ++t) {
            const int n = col0 + t * 16 + l15;
            // B 32x16 bf16, same contiguous k order as A: lane-half hi selects K 0..15 vs 16..31 of column n
            const uint4* Br = (const uint4*)(Bt + (size_t)n * 256 + k + hi * 16);
            Frag16 b;
            b.q[0] = Br[0];
            b.q[1] = Br[1];
            acc[t] = __builtin_amdgcn_wmma_f32_16x16x32_bf16(
                false, a.v, false, b.v, (short)0, acc[t], false, false);
        }
    }
    #pragma unroll
    for (int t = 0; t < 4; ++t) {
        const int col = col0 + t * 16 + l15;
        const float bv = bias[col];
        #pragma unroll
        for (int r = 0; r < 8; ++r) {
            const int mo = row0 + r + hi * 8;   // C layout: VGPR r -> rows r / r+8
            if (mo < rows) {
                float v = acc[t][r] + bv;
                if (OUT_F32) outF[(size_t)mo * 256 + col] = v;
                else         outH[(size_t)mo * 256 + col] = f2bf(v);
            }
        }
    }
}

// ---------------------------------------------------------------------------
// scores[m,h] = scale * dot(k[m,h,:], q[idx[m],h,:]); atomicMax into segmax.
__global__ void scores_kernel(const unsigned short* __restrict__ kb,
                              const unsigned short* __restrict__ qb,
                              const int* __restrict__ idx, float* __restrict__ scores,
                              unsigned* __restrict__ segmax, int M)
{
    const int wave = threadIdx.x >> 5, lane = threadIdx.x & 31;
    const int m = blockIdx.x * 8 + wave;
    if (m >= M) return;
    const int seg = idx[m];
    U4 kv, qv;
    kv.q = ((const uint4*)(kb + (size_t)m   * 256))[lane];
    qv.q = ((const uint4*)(qb + (size_t)seg * 256))[lane];
    float p = 0.0f;
    #pragma unroll
    for (int j = 0; j < 8; ++j) p += bf2f(kv.s[j]) * bf2f(qv.s[j]);
    p += __shfl_xor(p, 1);
    p += __shfl_xor(p, 2);
    if ((lane & 3) == 0) {
        const int h = lane >> 2;
        const float s = p * 0.1767766953f;   // 1/sqrt(32)
        scores[(size_t)m * 8 + h] = s;
        atomicMax(&segmax[(size_t)seg * 8 + h], ordf(s));
    }
}

// pooled[seg] += ew*v[m]; denom[seg,h] += ew
__global__ void pool_kernel(const unsigned short* __restrict__ vb,
                            const float* __restrict__ scores,
                            const unsigned* __restrict__ segmax,
                            const int* __restrict__ idx,
                            float* __restrict__ pooled, float* __restrict__ denom, int M)
{
    const int wave = threadIdx.x >> 5, lane = threadIdx.x & 31;
    const int m = blockIdx.x * 8 + wave;
    if (m >= M) return;
    const int seg = idx[m];
    const int h = lane >> 2;
    const float s  = scores[(size_t)m * 8 + h];
    const float mx = unordf(segmax[(size_t)seg * 8 + h]);
    const float ew = __expf(s - mx);
    U4 vv;
    vv.q = ((const uint4*)(vb + (size_t)m * 256))[lane];
    float* base = pooled + (size_t)seg * 256 + lane * 8;
    #pragma unroll
    for (int j = 0; j < 8; ++j) unsafeAtomicAdd(base + j, ew * bf2f(vv.s[j]));
    if ((lane & 3) == 0) unsafeAtomicAdd(&denom[(size_t)seg * 8 + h], ew);
}

// pooledb = bf16(pooled / max(denom, 1e-9))
__global__ void norm_kernel(const float* __restrict__ pooled, const float* __restrict__ denom,
                            unsigned short* __restrict__ pooledb, int S)
{
    long long i = (long long)blockIdx.x * blockDim.x + threadIdx.x;
    if (i >= (long long)S * 256) return;
    const int s = (int)(i >> 8);
    const int h = ((int)i & 255) >> 5;
    const float d = fmaxf(denom[(size_t)s * 8 + h], 1e-9f);
    pooledb[i] = f2bf(pooled[i] / d);
}

// ---------------------------------------------------------------------------
extern "C" void kernel_launch(void* const* d_in, const int* in_sizes, int n_in,
                              void* d_out, int out_size, void* d_ws, size_t ws_size,
                              hipStream_t stream)
{
    const float* enc = (const float*)d_in[0];
    const float* Wq  = (const float*)d_in[1];  const float* bq = (const float*)d_in[2];
    const float* Wk  = (const float*)d_in[3];  const float* bk = (const float*)d_in[4];
    const float* Wv  = (const float*)d_in[5];  const float* bv = (const float*)d_in[6];
    const float* Wo  = (const float*)d_in[7];  const float* bo = (const float*)d_in[8];
    const int* mapk  = (const int*)d_in[9];
    const int* mapv  = (const int*)d_in[10];
    const int* pk    = (const int*)d_in[11];
    const int* pv    = (const int*)d_in[12];

    const int M = in_sizes[9];    // 400000 mapping entries
    const int S = in_sizes[11];   // 100000 cfg nodes

    // --- carve workspace (256B-aligned blocks) ---
    char* p = (char*)d_ws;
    auto alloc = [&](size_t bytes) -> char* {
        char* r = p; p += (bytes + 255) & ~(size_t)255; return r;
    };
    unsigned short* xb      = (unsigned short*)alloc((size_t)M * 256 * 2);
    unsigned short* attnq   = (unsigned short*)alloc((size_t)S * 256 * 2);
    unsigned short* qb      = (unsigned short*)alloc((size_t)S * 256 * 2);
    unsigned short* kb      = (unsigned short*)alloc((size_t)M * 256 * 2);
    unsigned short* vb      = (unsigned short*)alloc((size_t)M * 256 * 2);
    unsigned short* pooledb = (unsigned short*)alloc((size_t)S * 256 * 2);
    unsigned short* Wqt     = (unsigned short*)alloc(65536 * 2);
    unsigned short* Wkt     = (unsigned short*)alloc(65536 * 2);
    unsigned short* Wvt     = (unsigned short*)alloc(65536 * 2);
    unsigned short* Wot     = (unsigned short*)alloc(65536 * 2);
    float*    scores = (float*)   alloc((size_t)M * 8 * 4);
    unsigned* segmax = (unsigned*)alloc((size_t)S * 8 * 4);
    float*    denom  = (float*)   alloc((size_t)S * 8 * 4);
    float*    pooled = (float*)   alloc((size_t)S * 256 * 4);

    // --- weights -> transposed bf16 ---
    wconv_kernel<<<256, 256, 0, stream>>>(Wq, Wqt);
    wconv_kernel<<<256, 256, 0, stream>>>(Wk, Wkt);
    wconv_kernel<<<256, 256, 0, stream>>>(Wv, Wvt);
    wconv_kernel<<<256, 256, 0, stream>>>(Wo, Wot);

    // --- init accumulators ---
    {
        long long total = (long long)S * 256;
        int blocks = (int)((total + 255) / 256);
        init_kernel<<<blocks, 256, 0, stream>>>(pooled, attnq, segmax, denom, S);
    }

    // --- gathers ---
    gather_x_kernel<<<M, 256, 0, stream>>>(enc, mapk, xb);
    scatter_q_kernel<<<S, 256, 0, stream>>>(enc, pk, pv, attnq);

    // --- projections (WMMA bf16, f32 accumulate) ---
    const int gS = (S + 15) / 16, gM = (M + 15) / 16;
    gemm256_bf16<0><<<gS, 128, 0, stream>>>(attnq, Wqt, bq, nullptr, qb, S);
    gemm256_bf16<0><<<gM, 128, 0, stream>>>(xb,    Wkt, bk, nullptr, kb, M);
    gemm256_bf16<0><<<gM, 128, 0, stream>>>(xb,    Wvt, bv, nullptr, vb, M);

    // --- segment softmax + pooling ---
    scores_kernel<<<(M + 7) / 8, 256, 0, stream>>>(kb, qb, mapv, scores, segmax, M);
    pool_kernel  <<<(M + 7) / 8, 256, 0, stream>>>(vb, scores, segmax, mapv, pooled, denom, M);
    norm_kernel  <<<(int)(((long long)S * 256 + 255) / 256), 256, 0, stream>>>(pooled, denom, pooledb, S);

    // --- output projection -> d_out (f32) ---
    gemm256_bf16<1><<<gS, 128, 0, stream>>>(pooledb, Wot, bo, (float*)d_out, nullptr, S);
}
